// COLA_64733747085779
// MI455X (gfx1250) — matrix operand order repaired and checked by reading, not compiled, not run on hardware
//
#include <hip/hip_runtime.h>
#include <hip/hip_bf16.h>
#include <math.h>

// ---------------------------------------------------------------------------
// CDNA5 (gfx1250) WMMA helpers — wave32, 16x16x32 f16 -> f32
// ---------------------------------------------------------------------------
typedef __attribute__((ext_vector_type(16))) _Float16 v16h;
typedef __attribute__((ext_vector_type(8)))  float    v8f;

static __device__ __forceinline__ v8f wmma_f32_f16(v16h a, v16h b, v8f c) {
  // 8 args: (neg_a, A, neg_b, B, c_mod, C, reuse_a, reuse_b)
  return __builtin_amdgcn_wmma_f32_16x16x32_f16(false, a, false, b, (short)0, c,
                                                false, false);
}

static __device__ __forceinline__ float sigf(float x) {
  return 1.f / (1.f + __expf(-x));
}
static __device__ __forceinline__ float tanh_fast(float x) {
  float e = __expf(2.f * x);
  return (e - 1.f) / (e + 1.f);
}

// Problem constants
#define BB 32
#define WW 16
#define MM 256
#define FF 16
#define HH 64
#define HALF 32
#define KC 10
#define NROWS (BB * MM)   // 8192

// ---------------------------------------------------------------------------
// Kernel 1: fused GRU over T=16. Each wave owns 16 sequences; hidden state
// lives in v8f WMMA accumulators. Weights are staged in LDS pre-swizzled to
// FRAGMENT-NATIVE layout: [tile][lane][16 contiguous halves], so every
// B-fragment is a single 32-byte vector LDS load (2x ds_load_b128).
// last_hid: (8192, 64)
// ---------------------------------------------------------------------------
__global__ void COLA_gru_kernel(const float* __restrict__ X,
                                const float* __restrict__ Wih,
                                const float* __restrict__ Whh,
                                const float* __restrict__ bih,
                                const float* __restrict__ bhh,
                                float* __restrict__ last_hid) {
  __shared__ v16h sWi[12 * 32];      // input-proj frags: [ntile][lane]  12 KB
  __shared__ v16h sWh[24 * 32];      // recurrent frags: [ntile*2+kc][lane] 24 KB
  __shared__ v16h sHF[4 * 2 * 32];   // per-wave h in A-frag layout [kc][lane] 8 KB

  _Float16* wi = (_Float16*)sWi;
  _Float16* wh = (_Float16*)sWh;
  const int tid = threadIdx.x;

  // Stage Wih^T (192x16, K padded to 32) into fragment layout.
  for (int idx = tid; idx < 12 * 512; idx += 128) {
    int ntile = idx >> 9, rem = idx & 511, l2 = rem >> 4, q = rem & 15;
    int n = ntile * 16 + (l2 & 15);
    int k = ((l2 & 16) ? 8 : 0) + (q & 7) + ((q & 8) ? 16 : 0);
    wi[idx] = (k < 16) ? (_Float16)Wih[n * 16 + k] : (_Float16)0.f;
  }
  // Stage Whh^T (192x64) into fragment layout (12 n-tiles x 2 k-chunks).
  for (int idx = tid; idx < 24 * 512; idx += 128) {
    int t2 = idx >> 9, rem = idx & 511, l2 = rem >> 4, q = rem & 15;
    int ntile = t2 >> 1, kc = t2 & 1;
    int n = ntile * 16 + (l2 & 15);
    int k = kc * 32 + ((l2 & 16) ? 8 : 0) + (q & 7) + ((q & 8) ? 16 : 0);
    wh[idx] = (_Float16)Whh[n * 64 + k];
  }
  __syncthreads();

  const int wave = tid >> 5;
  const int lane = tid & 31;
  const int ln15 = lane & 15;
  const int rowbase = blockIdx.x * 64 + wave * 16;
  v16h* hF = &sHF[wave * 64];
  _Float16* hFh = (_Float16*)hF;

  // per-lane gate biases (column = j*16 + ln15)
  float bir[4], biz[4], bin_[4], bhr[4], bhz[4], bhn[4];
#pragma unroll
  for (int j = 0; j < 4; ++j) {
    int col = j * 16 + ln15;
    bir[j]  = bih[col];        bhr[j] = bhh[col];
    biz[j]  = bih[64 + col];   bhz[j] = bhh[64 + col];
    bin_[j] = bih[128 + col];  bhn[j] = bhh[128 + col];
  }

  v8f h[4];
#pragma unroll
  for (int j = 0; j < 4; ++j) h[j] = {};

  const int seq = rowbase + ln15;        // A-fragment row (sequence id)
  const int bbi = seq >> 8;
  const int mmi = seq & 255;
  const int kbA = (lane & 16) ? 8 : 0;   // A-layout K base
  const int rb  = (lane & 16) ? 8 : 0;   // C-layout row base

  // precompute per-j scatter base for the C->A layout transpose of h
  int hbase[4];
#pragma unroll
  for (int j = 0; j < 4; ++j) {
    int c = j * 16 + ln15;               // hidden column this lane holds
    int chunk = c >> 5, cc = c & 31;
    int hi = (cc >> 3) & 1;
    int qq = (cc & 7) | (((cc >> 4) & 1) << 3);
    hbase[j] = chunk * 512 + hi * 256 + rb * 16 + qq;
  }

  for (int t = 0; t < WW; ++t) {
    // scatter h (C layout) into per-wave A-fragment layout buffer
#pragma unroll
    for (int j = 0; j < 4; ++j)
#pragma unroll
      for (int r = 0; r < 8; ++r)
        hFh[hbase[j] + r * 16] = (_Float16)h[j][r];
    __syncthreads();

    // A-fragments: x_t (K=16, zero-padded to 32) and h (K=64 -> 2 frags)
    v16h ax;
    const float* xp = X + (((long)(bbi * WW + t) * MM + mmi) << 4);
#pragma unroll
    for (int q = 0; q < 8; ++q) {
      ax[q]     = (_Float16)xp[kbA + q];
      ax[q + 8] = (_Float16)0.f;
    }
    v16h ah0 = hF[lane];
    v16h ah1 = hF[32 + lane];
    __syncthreads();

#pragma unroll
    for (int j = 0; j < 4; ++j) {
      v8f gir = {}, giz = {}, gin = {}, ghr = {}, ghz = {}, ghn = {};
      gir = wmma_f32_f16(ax, sWi[j * 32 + lane], gir);
      giz = wmma_f32_f16(ax, sWi[(4 + j) * 32 + lane], giz);
      gin = wmma_f32_f16(ax, sWi[(8 + j) * 32 + lane], gin);
      ghr = wmma_f32_f16(ah0, sWh[(j * 2 + 0) * 32 + lane], ghr);
      ghr = wmma_f32_f16(ah1, sWh[(j * 2 + 1) * 32 + lane], ghr);
      ghz = wmma_f32_f16(ah0, sWh[((4 + j) * 2 + 0) * 32 + lane], ghz);
      ghz = wmma_f32_f16(ah1, sWh[((4 + j) * 2 + 1) * 32 + lane], ghz);
      ghn = wmma_f32_f16(ah0, sWh[((8 + j) * 2 + 0) * 32 + lane], ghn);
      ghn = wmma_f32_f16(ah1, sWh[((8 + j) * 2 + 1) * 32 + lane], ghn);
#pragma unroll
      for (int r = 0; r < 8; ++r) {
        float rg = sigf((gir[r] + bir[j]) + (ghr[r] + bhr[j]));
        float zg = sigf((giz[r] + biz[j]) + (ghz[r] + bhz[j]));
        float ng = tanh_fast((gin[r] + bin_[j]) + rg * (ghn[r] + bhn[j]));
        h[j][r] = (1.f - zg) * ng + zg * h[j][r];
      }
    }
  }

  // write last hidden state (C layout -> global)
#pragma unroll
  for (int j = 0; j < 4; ++j)
#pragma unroll
    for (int r = 0; r < 8; ++r) {
      int n = rowbase + rb + r;
      last_hid[(long)n * HH + j * 16 + ln15] = h[j][r];
    }
}

// ---------------------------------------------------------------------------
// Kernel 2: generic batched WMMA GEMM.  C = act(A(MxK) * B(KxN) [+bias])
// block = 128 threads = 4 waves; wave -> one 16x16 C tile; all 4 waves share
// one 16-column B panel, staged per 32-k chunk into LDS in fragment layout.
// bT: B stored (N,K) transposed.  act: 0 none(+bias), 1 relu(+bias),
// 2 gated-adjacency epilogue: sig = sigmoid(acc + wb);
//     C = adj * sig + gate * (1 - sig)
// NOTE: launch grids must tile M exactly (all our shapes do) so control flow
// stays uniform across barriers and EXEC is all-ones at every WMMA.
// ---------------------------------------------------------------------------
__global__ void COLA_gemm_wmma(const float* __restrict__ A,
                               const float* __restrict__ B,
                               const float* __restrict__ bias,
                               float* __restrict__ C,
                               const float* __restrict__ adj,
                               const float* __restrict__ gate,
                               const float* __restrict__ wbp,
                               int M, int N, int K, int lda, int ldb, int ldc,
                               long sA, long sB, long sC, int bT, int act) {
  __shared__ v16h sBf[32];             // one 32x16 B chunk in fragment layout
  _Float16* sBh = (_Float16*)sBf;

  const int lane = threadIdx.x & 31;
  const int wave = threadIdx.x >> 5;
  const int row0 = (blockIdx.y * 4 + wave) * 16;
  const int n0 = blockIdx.x * 16;
  A += (long)blockIdx.z * sA;
  B += (long)blockIdx.z * sB;
  C += (long)blockIdx.z * sC;
  const float* gp = gate ? (gate + (long)blockIdx.z * sC) : nullptr;

  const int mA  = row0 + (lane & 15);
  const int kbA = (lane & 16) ? 8 : 0;
  const int nB  = n0 + (lane & 15);
  const bool mv = mA < M;
  const float* ap = A + (long)mA * lda;

  v8f acc = {};
  for (int k0 = 0; k0 < K; k0 += 32) {
    // cooperative stage of the B chunk into fragment-native LDS layout
#pragma unroll
    for (int e = 0; e < 4; ++e) {
      int idx = threadIdx.x * 4 + e;   // 0..511
      int l2 = idx >> 4, q = idx & 15;
      int n = n0 + (l2 & 15);
      int k = k0 + ((l2 & 16) ? 8 : 0) + (q & 7) + ((q & 8) ? 16 : 0);
      float v = 0.f;
      if (n < N && k < K) v = bT ? B[(long)n * ldb + k] : B[(long)k * ldb + n];
      sBh[idx] = (_Float16)v;
    }
    __syncthreads();

    if (k0 + 32 < K) __builtin_prefetch(ap + k0 + 32, 0, 3);

    v16h a;
#pragma unroll
    for (int q = 0; q < 8; ++q) {
      int k1 = k0 + kbA + q, k2 = k0 + kbA + 16 + q;
      a[q]     = (mv && k1 < K) ? (_Float16)ap[k1] : (_Float16)0.f;
      a[q + 8] = (mv && k2 < K) ? (_Float16)ap[k2] : (_Float16)0.f;
    }
    v16h b = sBf[lane];
    __syncthreads();
    acc = wmma_f32_f16(a, b, acc);
  }

  const bool nv = nB < N;
  if (!nv) return;
  const int rb = (lane & 16) ? 8 : 0;
  float bval = bias ? bias[nB] : 0.f;
#pragma unroll
  for (int r = 0; r < 8; ++r) {
    int m = row0 + rb + r;
    if (m >= M) continue;
    long idx = (long)m * ldc + nB;
    float v = acc[r];
    if (act == 1) {
      v = fmaxf(v + bval, 0.f);
    } else if (act == 2) {
      float s = sigf(v + wbp[0]);
      v = adj[idx] * s + gp[idx] * (1.f - s);
    } else {
      v += bval;
    }
    C[idx] = v;
  }
}

// ---------------------------------------------------------------------------
// Kernel 3: attention a[b,i,j] = elu(s1[b,j]+s2[b,i]+b1) . V + bv
// ---------------------------------------------------------------------------
__global__ void COLA_attn_kernel(const float* __restrict__ s1,
                                 const float* __restrict__ s2,
                                 const float* __restrict__ b1,
                                 const float* __restrict__ V,
                                 const float* __restrict__ bvp,
                                 float* __restrict__ amx) {
  const int i = blockIdx.x, b = blockIdx.y, j = threadIdx.x;
  __shared__ float s2s[HALF], b1s[HALF], Vs[HALF];
  if (threadIdx.x < HALF) {
    s2s[threadIdx.x] = s2[((long)(b << 8) + i) * HALF + threadIdx.x];
    b1s[threadIdx.x] = b1[threadIdx.x];
    Vs[threadIdx.x]  = V[threadIdx.x];
  }
  __syncthreads();
  const float* s1p = s1 + ((long)(b << 8) + j) * HALF;
  float acc = bvp[0];
#pragma unroll
  for (int k = 0; k < HALF; ++k) {
    float x = s1p[k] + s2s[k] + b1s[k];
    float e = x > 0.f ? x : (__expf(x) - 1.f);
    acc += e * Vs[k];
  }
  amx[(((long)b << 8) + i) * MM + j] = acc;
}

// ---------------------------------------------------------------------------
// Kernel 4: column normalization over axis=1 (i):  a /= max(||a[:,j]||, eps)
// ---------------------------------------------------------------------------
__global__ void COLA_norm_kernel(float* __restrict__ amx) {
  const int j = blockIdx.x, b = blockIdx.y, i = threadIdx.x;
  long idx = (((long)b << 8) + i) * MM + j;
  float v = amx[idx];
  __shared__ float red[256];
  red[i] = v * v;
  __syncthreads();
  for (int s = 128; s > 0; s >>= 1) {
    if (i < s) red[i] += red[i + s];
    __syncthreads();
  }
  float nrm = fmaxf(sqrtf(red[0]), 1e-12f);
  amx[idx] = v / nrm;
}

// ---------------------------------------------------------------------------
// Kernel 5: short + dilated long conv per node, ReLU, concat -> r_l (8192,30)
// ---------------------------------------------------------------------------
__global__ void COLA_conv_kernel(const float* __restrict__ X,
                                 const float* __restrict__ cw,
                                 const float* __restrict__ cb,
                                 const float* __restrict__ clw,
                                 const float* __restrict__ clb,
                                 float* __restrict__ r_l) {
  int tid = blockIdx.x * blockDim.x + threadIdx.x;
  if (tid >= NROWS * KC) return;
  int n = tid / KC, k = tid - n * KC;
  int b = n >> 8, m = n & 255;
  float a = cb[k], l0 = clb[k], l1 = clb[k];
  for (int f = 0; f < FF; ++f) {
    float xr[WW];
#pragma unroll
    for (int w = 0; w < WW; ++w)
      xr[w] = X[(((long)(b * WW + w) * MM + m) << 4) + f];
    const float* cwp = cw + (k * FF + f) * WW;
#pragma unroll
    for (int w = 0; w < WW; ++w) a += xr[w] * cwp[w];
    const float* clp = clw + (k * FF + f) * 8;
#pragma unroll
    for (int j = 0; j < 8; ++j) {
      l0 += xr[2 * j] * clp[j];
      l1 += xr[2 * j + 1] * clp[j];
    }
  }
  float* out = r_l + (long)n * 30 + k * 3;
  out[0] = fmaxf(a, 0.f);
  out[1] = fmaxf(l0, 0.f);
  out[2] = fmaxf(l1, 0.f);
}

// ---------------------------------------------------------------------------
// Kernel 6: head — logits, sigmoid, per-block BCE partial sums; final reduce
// ---------------------------------------------------------------------------
__global__ void COLA_head_kernel(const float* __restrict__ spat,
                                 const float* __restrict__ hid,
                                 const float* __restrict__ Y,
                                 const float* __restrict__ ow,
                                 const float* __restrict__ ob,
                                 float* __restrict__ probs,
                                 float* __restrict__ partial) {
  int n = blockIdx.x * 256 + threadIdx.x;
  float l = ob[0];
  const float* sp = spat + (long)n * 10;
#pragma unroll
  for (int k = 0; k < 10; ++k) l += sp[k] * ow[k];
  const float* hp = hid + (long)n * HH;
#pragma unroll
  for (int k = 0; k < HH; ++k) l += hp[k] * ow[10 + k];
  probs[n] = sigf(l);
  float y = Y[n];
  float term = fmaxf(l, 0.f) - l * y + log1pf(__expf(-fabsf(l)));
  __shared__ float red[256];
  red[threadIdx.x] = term;
  __syncthreads();
  for (int s = 128; s > 0; s >>= 1) {
    if (threadIdx.x < s) red[threadIdx.x] += red[threadIdx.x + s];
    __syncthreads();
  }
  if (threadIdx.x == 0) partial[blockIdx.x] = red[0];
}

__global__ void COLA_loss_final(const float* __restrict__ partial,
                                float* __restrict__ out0) {
  if (threadIdx.x == 0) {
    float s = 0.f;
    for (int i = 0; i < 32; ++i) s += partial[i];
    out0[0] = s * (1.f / (float)NROWS);
  }
}

// ---------------------------------------------------------------------------
// Launch
// ---------------------------------------------------------------------------
extern "C" void kernel_launch(void* const* d_in, const int* in_sizes, int n_in,
                              void* d_out, int out_size, void* d_ws, size_t ws_size,
                              hipStream_t stream) {
  (void)in_sizes; (void)n_in; (void)out_size; (void)ws_size;
  const float* X      = (const float*)d_in[0];
  const float* Y      = (const float*)d_in[1];
  const float* adj    = (const float*)d_in[2];
  const float* V      = (const float*)d_in[3];
  const float* bv     = (const float*)d_in[4];
  const float* W1     = (const float*)d_in[5];
  const float* b1     = (const float*)d_in[6];
  const float* W2     = (const float*)d_in[7];
  const float* Wb     = (const float*)d_in[8];
  const float* wb     = (const float*)d_in[9];
  const float* conv_w = (const float*)d_in[10];
  const float* conv_b = (const float*)d_in[11];
  const float* convl_w= (const float*)d_in[12];
  const float* convl_b= (const float*)d_in[13];
  const float* gc1_w  = (const float*)d_in[14];
  const float* gc1_b  = (const float*)d_in[15];
  const float* gc2_w  = (const float*)d_in[16];
  const float* gc2_b  = (const float*)d_in[17];
  const float* Wih    = (const float*)d_in[18];
  const float* Whh    = (const float*)d_in[19];
  const float* bih    = (const float*)d_in[20];
  const float* bhh    = (const float*)d_in[21];
  const float* out_w  = (const float*)d_in[22];
  const float* out_b  = (const float*)d_in[23];

  float* ws = (float*)d_ws;
  float* last_hid = ws;                       // 8192*64   = 524288
  float* s1       = ws + 524288;              // 8192*32   = 262144
  float* s2       = ws + 786432;              // 8192*32   = 262144
  float* amx      = ws + 1048576;             // 32*256*256= 2097152
  float* Amix     = ws + 3145728;             // 2097152
  float* r_l      = ws + 5242880;             // 8192*30   = 245760
  float* h1pre    = ws + 5488640;             // 524288
  float* x1       = ws + 6012928;             // 524288
  float* x2pre    = ws + 6537216;             // 8192*10   = 81920
  float* spat     = ws + 6619136;             // 81920
  float* partial  = ws + 6701056;             // 32

  float* out = (float*)d_out;                 // [loss, probs(8192)]

  // 1) fused GRU (WMMA recurrence, fragment-native weights in LDS)
  COLA_gru_kernel<<<128, 128, 0, stream>>>(X, Wih, Whh, bih, bhh, last_hid);

  // 2) s1 = last_hid @ W1^T, s2 = last_hid @ W2^T   (M=8192, N=32, K=64)
  COLA_gemm_wmma<<<dim3(2, 128, 1), 128, 0, stream>>>(
      last_hid, W1, nullptr, s1, nullptr, nullptr, nullptr,
      NROWS, HALF, HH, HH, HH, HALF, 0, 0, 0, /*bT=*/1, /*act=*/0);
  COLA_gemm_wmma<<<dim3(2, 128, 1), 128, 0, stream>>>(
      last_hid, W2, nullptr, s2, nullptr, nullptr, nullptr,
      NROWS, HALF, HH, HH, HH, HALF, 0, 0, 0, 1, 0);

  // 3) bilinear attention + 4) column norm
  COLA_attn_kernel<<<dim3(MM, BB), MM, 0, stream>>>(s1, s2, b1, V, bv, amx);
  COLA_norm_kernel<<<dim3(MM, BB), MM, 0, stream>>>(amx);

  // 5) short + long convs -> r_l
  COLA_conv_kernel<<<(NROWS * KC + 255) / 256, 256, 0, stream>>>(
      X, conv_w, conv_b, convl_w, convl_b, r_l);

  // 6) gated adjacency: Amix = adj*sig(amx@Wb + wb) + amx*(1-sig)
  COLA_gemm_wmma<<<dim3(16, 4, BB), 128, 0, stream>>>(
      amx, Wb, nullptr, Amix, adj, amx, wb,
      MM, MM, MM, MM, MM, MM,
      (long)MM * MM, 0, (long)MM * MM, 0, /*act=*/2);

  // 7) h1pre = r_l @ gc1_w      (M=8192, N=64, K=30)
  COLA_gemm_wmma<<<dim3(4, 128, 1), 128, 0, stream>>>(
      r_l, gc1_w, nullptr, h1pre, nullptr, nullptr, nullptr,
      NROWS, HH, 30, 30, HH, HH, 0, 0, 0, 0, 0);

  // 8) x1 = relu(Amix @ h1pre + gc1_b)   batched 32 x (256x64x256)
  COLA_gemm_wmma<<<dim3(4, 4, BB), 128, 0, stream>>>(
      Amix, h1pre, gc1_b, x1, nullptr, nullptr, nullptr,
      MM, HH, MM, MM, HH, HH,
      (long)MM * MM, (long)MM * HH, (long)MM * HH, 0, /*act=*/1);

  // 9) x2pre = x1 @ gc2_w       (M=8192, N=10, K=64)
  COLA_gemm_wmma<<<dim3(1, 128, 1), 128, 0, stream>>>(
      x1, gc2_w, nullptr, x2pre, nullptr, nullptr, nullptr,
      NROWS, 10, HH, HH, 10, 10, 0, 0, 0, 0, 0);

  // 10) spat = relu(Amix @ x2pre + gc2_b)  batched 32 x (256x10x256)
  COLA_gemm_wmma<<<dim3(1, 4, BB), 128, 0, stream>>>(
      Amix, x2pre, gc2_b, spat, nullptr, nullptr, nullptr,
      MM, 10, MM, MM, 10, 10,
      (long)MM * MM, (long)MM * 10, (long)MM * 10, 0, /*act=*/1);

  // 11) head: logits -> probs + BCE partials; 12) final loss reduce
  COLA_head_kernel<<<32, 256, 0, stream>>>(spat, last_hid, Y, out_w, out_b,
                                           out + 1, partial);
  COLA_loss_final<<<1, 32, 0, stream>>>(partial, out);
}
